// MultiHeadAttention_26508538151175
// MI455X (gfx1250) — matrix operand order repaired and checked by reading, not compiled, not run on hardware
//
#include <hip/hip_runtime.h>
#include <hip/hip_bf16.h>

// ---------------- CDNA5 (gfx1250) types ----------------
typedef __bf16 bf16;
typedef __attribute__((ext_vector_type(16))) __bf16 v16bf;
typedef __attribute__((ext_vector_type(8)))  __bf16 v8bf;
typedef __attribute__((ext_vector_type(8)))  float  v8f;
typedef int v4i_b __attribute__((vector_size(16)));   // b128 payload type for async-LDS builtin

// Problem constants (from reference)
#define BATCH 2
#define HEADS 8
#define SEQ   4096
#define DMODEL 512
#define DK    64
#define MROWS (BATCH * SEQ)          // 8192

// Q pre-scale: (1/sqrt(64)) * log2(e)  -> scores arrive in log2-domain
#define QSCALE (0.125f * 1.44269504088896340736f)

// ---------------- CDNA5 async global->LDS path (ASYNCcnt) ----------------
#if defined(__AMDGCN__) && __has_builtin(__builtin_amdgcn_global_load_async_to_lds_b128)
#define HAVE_ASYNC_LDS 1
#else
#define HAVE_ASYNC_LDS 0
#endif

__device__ __forceinline__ void async_copy16(const bf16* gsrc, bf16* ldst) {
#if HAVE_ASYNC_LDS
  // signature (probe-discovered): (int4* global_src, int4* lds_dst, imm offset, imm cpol)
  __builtin_amdgcn_global_load_async_to_lds_b128(
      (v4i_b*)gsrc, (v4i_b*)ldst, 0, 0);
#else
  *(v8bf*)ldst = *(const v8bf*)gsrc;
#endif
}

__device__ __forceinline__ void wait_async() {
#if HAVE_ASYNC_LDS
#if __has_builtin(__builtin_amdgcn_s_wait_asynccnt)
  __builtin_amdgcn_s_wait_asynccnt(0);
#else
  asm volatile("s_wait_asynccnt 0x0" ::: "memory");
#endif
#endif
}

__device__ __forceinline__ float fast_exp2(float x) {
#if __has_builtin(__builtin_amdgcn_exp2f)
  return __builtin_amdgcn_exp2f(x);   // raw v_exp_f32
#else
  return exp2f(x);
#endif
}

__device__ __forceinline__ v8f wmma_bf16(v16bf a, v16bf b, v8f c) {
  // D = A(16x32) * B(32x16) + C(16x16 f32)
  return __builtin_amdgcn_wmma_f32_16x16x32_bf16(
      false, a, false, b, (short)0, c, false, false);
}

// ---- A-fragment (16x32 bf16), ISA 7.12.2 layout:
// lane L: M = L%16 ; lanes 0-15 hold K = e (e<8) and 16+e ; lanes 16-31 hold K = 8+e and 24+e
__device__ __forceinline__ v16bf load_a_frag(const bf16* base, int ld) {
  const int lane = threadIdx.x & 31;
  const int m  = lane & 15;
  const int k0 = (lane & 16) ? 8 : 0;
  const bf16* row = base + (size_t)m * ld + k0;
  v8bf lo = *(const v8bf*)(row);        // K = k0 .. k0+7
  v8bf hi = *(const v8bf*)(row + 16);   // K = 16+k0 .. 16+k0+7
  v16bf r;
#pragma unroll
  for (int i = 0; i < 8; ++i) { r[i] = lo[i]; r[i + 8] = hi[i]; }
  return r;
}

// ---- B-fragment (32x16 bf16) from [n][k] storage (k contiguous):
// lane L: N = L%16 ; lanes 0-15 hold K = e (e<8) and 16+e ; lanes 16-31 hold K = 8+e and 24+e
__device__ __forceinline__ v16bf load_b_frag_nk(const bf16* base, int ld) {
  const int lane = threadIdx.x & 31;
  const int n  = lane & 15;
  const int k0 = (lane & 16) ? 8 : 0;
  const bf16* row = base + (size_t)n * ld + k0;
  v8bf lo = *(const v8bf*)(row);
  v8bf hi = *(const v8bf*)(row + 16);
  v16bf r;
#pragma unroll
  for (int i = 0; i < 8; ++i) { r[i] = lo[i]; r[i + 8] = hi[i]; }
  return r;
}

__device__ __forceinline__ void stage_f32_16(const float* src, bf16* dst) {
  v8bf t0, t1;
#pragma unroll
  for (int j = 0; j < 8; ++j) { t0[j] = (bf16)src[j]; t1[j] = (bf16)src[j + 8]; }
  *(v8bf*)dst       = t0;
  *(v8bf*)(dst + 8) = t1;
}

// =====================================================================
// Kernel 1/3: out = oscale * (X @ W^T + bias). 64x64 tile per WG, 4 waves,
// each wave computes a 16x64 strip with 4 accumulating WMMAs per K-chunk.
//   IN_BF16=false, HEAD_SPLIT=true : fp32 X, bf16 out in [B,H,S,64] (+bias)
//   IN_BF16=true,  HEAD_SPLIT=false: bf16 X, fp32 out in [M,512] (no bias)
// =====================================================================
template <bool IN_BF16, bool HEAD_SPLIT>
__global__ __launch_bounds__(128) void gemm_proj(const void* __restrict__ xin,
                                                 const float* __restrict__ W,
                                                 const float* __restrict__ bias,
                                                 void* __restrict__ outv,
                                                 float oscale) {
  constexpr int LD = 40;  // 32 bf16 + pad (80B rows -> conflict-free strides)
  __shared__ __align__(16) bf16 As[64 * LD];
  __shared__ __align__(16) bf16 Bs[64 * LD];

  const int tid  = threadIdx.x;
  const int wave = tid >> 5;
  const int lane = tid & 31;
  const int m0 = blockIdx.x * 64;
  const int n0 = blockIdx.y * 64;
  const int lrow = tid >> 1;            // 0..63
  const int lcol = (tid & 1) * 16;      // 0 or 16

  v8f acc[4] = {{0}, {0}, {0}, {0}};

  for (int k0 = 0; k0 < DMODEL; k0 += 32) {
    // stage A tile (64 x 32) as bf16
    if (IN_BF16) {
      const bf16* src = (const bf16*)xin + (size_t)(m0 + lrow) * DMODEL + k0 + lcol;
      bf16* dst = &As[lrow * LD + lcol];
      async_copy16(src, dst);           // ASYNCcnt path on CDNA5
      async_copy16(src + 8, dst + 8);
    } else {
      const float* src = (const float*)xin + (size_t)(m0 + lrow) * DMODEL + k0 + lcol;
      stage_f32_16(src, &As[lrow * LD + lcol]);
    }
    // stage B tile: rows n of W (X @ W^T  =>  B[k][n] = W[n][k], k contiguous)
    {
      const float* src = W + (size_t)(n0 + lrow) * DMODEL + k0 + lcol;
      stage_f32_16(src, &Bs[lrow * LD + lcol]);
    }
    wait_async();
    __syncthreads();

    v16bf a = load_a_frag(As + wave * 16 * LD, LD);
#pragma unroll
    for (int nt = 0; nt < 4; ++nt) {
      v16bf b = load_b_frag_nk(Bs + nt * 16 * LD, LD);
      acc[nt] = wmma_bf16(a, b, acc[nt]);
    }
    __syncthreads();
  }

  // epilogue: C layout -> per lane: col N fixed (= lane%16), rows vary by VGPR
#pragma unroll
  for (int nt = 0; nt < 4; ++nt) {
    const int n = n0 + nt * 16 + (lane & 15);
    const float bn = HEAD_SPLIT ? bias[n] : 0.0f;
#pragma unroll
    for (int i = 0; i < 8; ++i) {
      const int m = m0 + wave * 16 + i + ((lane & 16) ? 8 : 0);
      const float val = (acc[nt][i] + bn) * oscale;
      if (HEAD_SPLIT) {
        const int b  = m >> 12;          // m / SEQ
        const int s  = m & (SEQ - 1);
        const int h  = n >> 6;
        const int dk = n & 63;
        ((bf16*)outv)[(((size_t)(b * HEADS + h) * SEQ + s) << 6) + dk] = (bf16)val;
      } else {
        ((float*)outv)[(size_t)m * DMODEL + n] = val;
      }
    }
  }
}

// =====================================================================
// Kernel 2: flash attention. Grid (SEQ/64, B*H). 4 waves/WG; wave owns 16
// query rows; all waves stream the same 64-key K/V blocks through LDS.
// Q was pre-scaled by 0.125*log2(e), so scores are already in log2 units
// and the whole online softmax runs on raw v_exp_f32 (exp2).
// =====================================================================
__global__ __launch_bounds__(128) void attn_kernel(const bf16* __restrict__ Q,
                                                   const bf16* __restrict__ K,
                                                   const bf16* __restrict__ V,
                                                   const int* __restrict__ mask,
                                                   bf16* __restrict__ X) {
  constexpr int KB   = 64;  // keys per iteration
  constexpr int LDKS = 72;  // [key][dk]: 64 bf16 + pad
  constexpr int LDVS = 72;  // [dk][key] (transposed): 64 bf16 + pad
  constexpr int LDPS = 72;  // per-wave P rows: 64 bf16 + pad
  __shared__ __align__(16) bf16 Ks[KB * LDKS];
  __shared__ __align__(16) bf16 Vs[DK * LDVS];
  __shared__ __align__(16) bf16 Ps[4 * 16 * LDPS];

  const int tid  = threadIdx.x;
  const int wave = tid >> 5;
  const int lane = tid & 31;
  const int bh = blockIdx.y;           // b*H + h
  const int b  = bh >> 3;
  const int h  = bh & 7;
  const int q0 = blockIdx.x * 64;
  const size_t baseQK = (size_t)bh * SEQ * DK;

  // persistent Q fragments for this wave's 16 rows (dk 0..31 and 32..63)
  const bf16* qbase = Q + baseQK + (size_t)(q0 + wave * 16) * DK;
  const v16bf qa0 = load_a_frag(qbase, DK);
  const v16bf qa1 = load_a_frag(qbase + 32, DK);

  float mrow[8], lrow[8];
  v8f o[4] = {{0}, {0}, {0}, {0}};
#pragma unroll
  for (int i = 0; i < 8; ++i) { mrow[i] = -1e30f; lrow[i] = 0.0f; }

  const int col  = lane & 15;
  const int rofs = (lane & 16) ? 8 : 0;
  bf16* Pw = Ps + wave * 16 * LDPS;

  for (int kb = 0; kb < SEQ; kb += KB) {
    // ---- cooperative K/V stage: thread -> key row (tid&63), 32 dk values ----
    {
      const int key = tid & 63;
      const int ch  = (tid >> 6) * 32;  // 0 or 32
      const bf16* kg = K + baseQK + (size_t)(kb + key) * DK + ch;
      bf16* kd = &Ks[key * LDKS + ch];
      async_copy16(kg, kd);             // K block: pure byte-move -> async path
      async_copy16(kg + 8, kd + 8);
      async_copy16(kg + 16, kd + 16);
      async_copy16(kg + 24, kd + 24);
      const bf16* vg = V + baseQK + (size_t)(kb + key) * DK + ch;
      v8bf v0 = *(const v8bf*)vg,        v1 = *(const v8bf*)(vg + 8);
      v8bf v2 = *(const v8bf*)(vg + 16), v3 = *(const v8bf*)(vg + 24);
#pragma unroll
      for (int j = 0; j < 8; ++j) {     // transpose V into [dk][key]
        Vs[(ch + j) * LDVS + key]      = v0[j];
        Vs[(ch + 8 + j) * LDVS + key]  = v1[j];
        Vs[(ch + 16 + j) * LDVS + key] = v2[j];
        Vs[(ch + 24 + j) * LDVS + key] = v3[j];
      }
      if (kb + KB < SEQ) {  // global_prefetch_b8 of next block
        __builtin_prefetch(kg + KB * DK, 0, 1);
        __builtin_prefetch(vg + KB * DK, 0, 1);
      }
    }
    wait_async();
    __syncthreads();

    // ---- scores: S(16x64) = Q(16x64) . K^T, four 16x16 tiles, K-dim 64 ----
    v8f s[4] = {{0}, {0}, {0}, {0}};
#pragma unroll
    for (int t = 0; t < 4; ++t) {
      v16bf bk0 = load_b_frag_nk(Ks + t * 16 * LDKS, LDKS);       // dk 0-31
      v16bf bk1 = load_b_frag_nk(Ks + t * 16 * LDKS + 32, LDKS);  // dk 32-63
      s[t] = wmma_bf16(qa0, bk0, s[t]);
      s[t] = wmma_bf16(qa1, bk1, s[t]);
    }

    // ---- online softmax in log2 domain (row lives in one 16-lane half) ----
#pragma unroll
    for (int i = 0; i < 8; ++i) {
      const int qrow = q0 + wave * 16 + i + rofs;
      const size_t mbase = ((size_t)b * SEQ + qrow) * (size_t)SEQ + kb + col;
      float x[4];
#pragma unroll
      for (int t = 0; t < 4; ++t) {
        x[t] = s[t][i];                       // already scaled by 0.125*log2e
        if (mask[mbase + t * 16] == 0) x[t] = -1e9f;
      }
      float mx = fmaxf(fmaxf(x[0], x[1]), fmaxf(x[2], x[3]));
#pragma unroll
      for (int off = 1; off < 16; off <<= 1)
        mx = fmaxf(mx, __shfl_xor(mx, off, 32));
      const float mnew = fmaxf(mrow[i], mx);
      const float corr = fast_exp2(mrow[i] - mnew);
      float p[4], rs = 0.0f;
#pragma unroll
      for (int t = 0; t < 4; ++t) { p[t] = fast_exp2(x[t] - mnew); rs += p[t]; }
#pragma unroll
      for (int off = 1; off < 16; off <<= 1)
        rs += __shfl_xor(rs, off, 32);
      lrow[i] = lrow[i] * corr + rs;
      mrow[i] = mnew;
#pragma unroll
      for (int dt = 0; dt < 4; ++dt) o[dt][i] *= corr;

      // C-layout -> A-layout transpose of P via per-wave LDS scratch
#pragma unroll
      for (int t = 0; t < 4; ++t)
        Pw[(i + rofs) * LDPS + t * 16 + col] = (bf16)p[t];
    }

    // ---- O += P(16x64) . V(64x64) ----
    {
      v16bf pa0 = load_a_frag(Pw, LDPS);        // keys 0-31
      v16bf pa1 = load_a_frag(Pw + 32, LDPS);   // keys 32-63
#pragma unroll
      for (int dt = 0; dt < 4; ++dt) {
        v16bf vb0 = load_b_frag_nk(Vs + dt * 16 * LDVS, LDVS);
        v16bf vb1 = load_b_frag_nk(Vs + dt * 16 * LDVS + 32, LDVS);
        o[dt] = wmma_bf16(pa0, vb0, o[dt]);
        o[dt] = wmma_bf16(pa1, vb1, o[dt]);
      }
    }
    __syncthreads();
  }

  // ---- finalize + store merged-head bf16 [B, S, H*64] ----
#pragma unroll
  for (int dt = 0; dt < 4; ++dt) {
    const int dk = dt * 16 + (lane & 15);
#pragma unroll
    for (int i = 0; i < 8; ++i) {
      const int s = q0 + wave * 16 + i + rofs;
      const float val = o[dt][i] / lrow[i];
      X[((size_t)(b * SEQ + s) << 9) + h * 64 + dk] = (bf16)val;
    }
  }
}

// =====================================================================
extern "C" void kernel_launch(void* const* d_in, const int* in_sizes, int n_in,
                              void* d_out, int out_size, void* d_ws, size_t ws_size,
                              hipStream_t stream) {
  (void)in_sizes; (void)n_in; (void)out_size; (void)ws_size;
  const float* query = (const float*)d_in[0];
  const float* key   = (const float*)d_in[1];
  const float* value = (const float*)d_in[2];
  const int*   mask  = (const int*)d_in[3];
  const float* Wq = (const float*)d_in[4];
  const float* bq = (const float*)d_in[5];
  const float* Wk = (const float*)d_in[6];
  const float* bk = (const float*)d_in[7];
  const float* Wv = (const float*)d_in[8];
  const float* bv = (const float*)d_in[9];
  const float* Wo = (const float*)d_in[10];

  const size_t QKV_ELEMS = (size_t)BATCH * HEADS * SEQ * DK;  // 4,194,304
  bf16* Qw = (bf16*)d_ws;
  bf16* Kw = Qw + QKV_ELEMS;
  bf16* Vw = Kw + QKV_ELEMS;
  bf16* Xw = Vw + QKV_ELEMS;   // merged-head attention output (bf16)

  dim3 blk(128, 1, 1);
  dim3 gproj(MROWS / 64, DMODEL / 64, 1);  // (128, 8)

  gemm_proj<false, true><<<gproj, blk, 0, stream>>>(query, Wq, bq, Qw, QSCALE);
  gemm_proj<false, true><<<gproj, blk, 0, stream>>>(key,   Wk, bk, Kw, 1.0f);
  gemm_proj<false, true><<<gproj, blk, 0, stream>>>(value, Wv, bv, Vw, 1.0f);

  attn_kernel<<<dim3(SEQ / 64, BATCH * HEADS, 1), blk, 0, stream>>>(Qw, Kw, Vw, mask, Xw);

  gemm_proj<true, false><<<gproj, blk, 0, stream>>>(Xw, Wo, nullptr, (float*)d_out, 1.0f);
}
